// GroupQueryAttention_3332894621882
// MI455X (gfx1250) — compile-verified, matrix operand order checked by reading, not run on hardware
//
#include <hip/hip_runtime.h>
#include <hip/hip_bf16.h>

// ---------------------------------------------------------------------------
// GQA attention block on gfx1250 (MI455X), all matmuls via v_wmma_f32_16x16x32_bf16
//   Pass 0: one-shot fp32 -> bf16 conversion of x and weights (GQA-compacted K/V)
//   Pass 1: Q/K/V projections as pure-bf16 WMMA GEMMs
//   Pass 2: causal GQA flash attention (WMMA for QK^T and PV, online softmax)
//   Pass 3: output projection WMMA GEMM + bias, fp32 out
// ---------------------------------------------------------------------------

#define EMB 2048
#define NH  16
#define NG  4
#define GSZ 4          // NH / NG
#define HD  128        // EMB / NH
#define BB  2
#define TT  2048

typedef __attribute__((ext_vector_type(16))) __bf16 v16bf;
typedef __attribute__((ext_vector_type(8)))  float  v8f;

// ---------------------------------------------------------------------------
// Row-gather fp32 -> bf16 conversion.
// dst row n (length K) = src row ((n/HD)*headStride + n%HD).
// headStride == HD       -> identity copy/convert
// headStride == GSZ*HD   -> keep only head 0 of each KV group (GQA compaction)
// One block per dst row; each thread converts a contiguous 8-element chunk so
// the backend emits b128 load + packed cvt + b128 store.
// ---------------------------------------------------------------------------
__global__ __launch_bounds__(256) void cvt_rows_f32_bf16(
    const float* __restrict__ src, __bf16* __restrict__ dst,
    int K, int headStride) {
  const int n = blockIdx.x;
  const int srow = (n / HD) * headStride + (n % HD);
  const float* s = src + (size_t)srow * K;
  __bf16* d = dst + (size_t)n * K;
  for (int k = threadIdx.x * 8; k < K; k += blockDim.x * 8) {
#pragma unroll
    for (int j = 0; j < 8; ++j) d[k + j] = (__bf16)s[k + j];
  }
}

// ---------------------------------------------------------------------------
// Pure-bf16 GEMM: C[M,N] = A[M,K] @ B[N,K]^T  (both row-major bf16).
// Block: 256 threads = 8 waves; block tile 128(M) x 64(N); wave tile 16 x 64.
// A-fragment: lane = row (lane&15), k = k0 + 8*(lane>>4) + {0..7, 16..23} — two 8-elem loads.
// B-fragment: lane = col n, same K layout — contiguous loads of B's row n.
// ---------------------------------------------------------------------------
template <typename OType>
__global__ __launch_bounds__(256) void wmma_gemm_bf16(
    const __bf16* __restrict__ A, const __bf16* __restrict__ Bw,
    OType* __restrict__ C, const float* __restrict__ bias,
    int M, int N, int K) {
  const int lane = threadIdx.x & 31;
  const int wave = threadIdx.x >> 5;
  const int n16  = lane & 15;
  const int hlf  = lane >> 4;
  const int kHalf = hlf * 8;

  const int mBase = blockIdx.y * 128 + wave * 16;
  const int nBase = blockIdx.x * 64;

  v8f acc[4] = {v8f{}, v8f{}, v8f{}, v8f{}};

  const __bf16* arow = A + (size_t)(mBase + n16) * K;
  const __bf16* brows[4];
#pragma unroll
  for (int nt = 0; nt < 4; ++nt)
    brows[nt] = Bw + (size_t)(nBase + nt * 16 + n16) * K;

  for (int k0 = 0; k0 < K; k0 += 32) {
    v16bf a;
#pragma unroll
    for (int j = 0; j < 8; ++j) {
      a[j]     = arow[k0 + kHalf + j];
      a[8 + j] = arow[k0 + kHalf + 16 + j];
    }
#pragma unroll
    for (int nt = 0; nt < 4; ++nt) {
      v16bf b;
#pragma unroll
      for (int j = 0; j < 8; ++j) {
        b[j]     = brows[nt][k0 + kHalf + j];
        b[8 + j] = brows[nt][k0 + kHalf + 16 + j];
      }
      acc[nt] = __builtin_amdgcn_wmma_f32_16x16x32_bf16(
          false, a, false, b, (short)0, acc[nt], false, false);
    }
  }

  // C/D layout: element j of v8f -> row (j + 8*hlf), lane&15 -> col
#pragma unroll
  for (int nt = 0; nt < 4; ++nt) {
#pragma unroll
    for (int j = 0; j < 8; ++j) {
      int r = mBase + 8 * hlf + j;
      int c = nBase + nt * 16 + n16;
      float v = acc[nt][j];
      if (bias) v += bias[c];
      C[(size_t)r * N + c] = (OType)v;
    }
  }
}

// ---------------------------------------------------------------------------
// Flash attention, causal, GQA (NG kv heads shared by GSZ query heads each).
// Q: (B,T,EMB) bf16, col = h*HD + hd
// Kc/Vc: (B,T,NG*HD) bf16, col = g*HD + hd   (head 0 of each group)
// O: (B,T,EMB) bf16
// Grid: (T/128, B*NH); block 256 = 8 waves; wave owns 16 query rows.
// ---------------------------------------------------------------------------
__global__ __launch_bounds__(256) void flash_attn_gqa(
    const __bf16* __restrict__ Q, const __bf16* __restrict__ Kc,
    const __bf16* __restrict__ Vc, __bf16* __restrict__ O) {
  __shared__ __bf16 plds[8][16][72];  // per-wave P tile (16 x 64, padded)

  const int lane = threadIdx.x & 31;
  const int wave = threadIdx.x >> 5;
  const int n16  = lane & 15;
  const int hlf  = lane >> 4;
  const int kHalf = hlf * 8;

  const int bh = blockIdx.y;
  const int b = bh / NH, h = bh % NH, g = h / GSZ;
  const int qBase = blockIdx.x * 128 + wave * 16;
  const int kvld = NG * HD;
  const float scale = 0.088388347648318447f;  // 1/sqrt(128)

  // Preload Q A-fragments: 4 k-steps over HD=128. A row = lane&15.
  v16bf qf[4];
  const __bf16* qrow = Q + ((size_t)(b * TT + qBase + n16)) * EMB + h * HD;
#pragma unroll
  for (int ks = 0; ks < 4; ++ks) {
#pragma unroll
    for (int j = 0; j < 8; ++j) {
      qf[ks][j]     = qrow[ks * 32 + kHalf + j];
      qf[ks][8 + j] = qrow[ks * 32 + kHalf + 16 + j];
    }
  }

  v8f of[8] = {v8f{}, v8f{}, v8f{}, v8f{}, v8f{}, v8f{}, v8f{}, v8f{}};
  float mstat[8], lstat[8];
#pragma unroll
  for (int j = 0; j < 8; ++j) { mstat[j] = -1e30f; lstat[j] = 0.0f; }

  const int kcEnd = (qBase + 15) / 64 + 1;  // chunks of 64 keys, causal bound
  for (int kc = 0; kc < kcEnd; ++kc) {
    const int kcBase = kc * 64;

    // ---- S = Q K^T for this 16x64 stripe -------------------------------
    v8f s[4];
#pragma unroll
    for (int nt = 0; nt < 4; ++nt) {
      v8f acc = {};
      const int key = kcBase + nt * 16 + n16;
      const __bf16* krow = Kc + ((size_t)(b * TT + key)) * kvld + g * HD;
      __builtin_prefetch(krow + 64 * kvld, 0, 1);  // global_prefetch next chunk
#pragma unroll
      for (int ks = 0; ks < 4; ++ks) {
        v16bf kb;
#pragma unroll
        for (int j = 0; j < 8; ++j) {
          kb[j]     = krow[ks * 32 + kHalf + j];
          kb[8 + j] = krow[ks * 32 + kHalf + 16 + j];
        }
        acc = __builtin_amdgcn_wmma_f32_16x16x32_bf16(
            false, qf[ks], false, kb, (short)0, acc, false, false);
      }
      s[nt] = acc;
    }

    // ---- scale, causal mask, online softmax ----------------------------
    float rowmax[8];
#pragma unroll
    for (int j = 0; j < 8; ++j) rowmax[j] = -1e30f;
#pragma unroll
    for (int nt = 0; nt < 4; ++nt) {
      const int kj = kcBase + nt * 16 + n16;
#pragma unroll
      for (int j = 0; j < 8; ++j) {
        const int qi = qBase + 8 * hlf + j;
        float v = s[nt][j] * scale;
        if (kj > qi) v = -1e30f;
        s[nt][j] = v;
        rowmax[j] = fmaxf(rowmax[j], v);
      }
    }
#pragma unroll
    for (int m = 8; m >= 1; m >>= 1)
#pragma unroll
      for (int j = 0; j < 8; ++j)
        rowmax[j] = fmaxf(rowmax[j], __shfl_xor(rowmax[j], m, 32));

    float fac[8], rs[8];
#pragma unroll
    for (int j = 0; j < 8; ++j) {
      float nm = fmaxf(mstat[j], rowmax[j]);
      fac[j] = __expf(mstat[j] - nm);
      mstat[j] = nm;
      rs[j] = 0.0f;
    }
#pragma unroll
    for (int nt = 0; nt < 4; ++nt)
#pragma unroll
      for (int j = 0; j < 8; ++j) {
        float p = __expf(s[nt][j] - mstat[j]);
        s[nt][j] = p;
        rs[j] += p;
      }
#pragma unroll
    for (int m = 8; m >= 1; m >>= 1)
#pragma unroll
      for (int j = 0; j < 8; ++j) rs[j] += __shfl_xor(rs[j], m, 32);
#pragma unroll
    for (int j = 0; j < 8; ++j) lstat[j] = lstat[j] * fac[j] + rs[j];
#pragma unroll
    for (int nt2 = 0; nt2 < 8; ++nt2)
#pragma unroll
      for (int j = 0; j < 8; ++j) of[nt2][j] *= fac[j];

    // ---- stage P in LDS to re-shape C-fragment -> A-fragment -----------
#pragma unroll
    for (int nt = 0; nt < 4; ++nt)
#pragma unroll
      for (int j = 0; j < 8; ++j)
        plds[wave][8 * hlf + j][nt * 16 + n16] = (__bf16)s[nt][j];
    // wave-private LDS region; split-counter wait instead of a barrier
    asm volatile("s_wait_dscnt 0" ::: "memory");

    // ---- O += P V ------------------------------------------------------
#pragma unroll
    for (int ks2 = 0; ks2 < 2; ++ks2) {
      const int kk = ks2 * 32;
      v16bf pa;
#pragma unroll
      for (int j = 0; j < 8; ++j) {
        pa[j]     = plds[wave][n16][kk + kHalf + j];
        pa[8 + j] = plds[wave][n16][kk + kHalf + 16 + j];
      }
#pragma unroll
      for (int nt2 = 0; nt2 < 8; ++nt2) {
        v16bf vb;
#pragma unroll
        for (int j = 0; j < 8; ++j) {
          int key0 = kcBase + kk + kHalf + j;
          int key1 = kcBase + kk + kHalf + 16 + j;
          vb[j]     = Vc[((size_t)(b * TT + key0)) * kvld + g * HD + nt2 * 16 + n16];
          vb[8 + j] = Vc[((size_t)(b * TT + key1)) * kvld + g * HD + nt2 * 16 + n16];
        }
        of[nt2] = __builtin_amdgcn_wmma_f32_16x16x32_bf16(
            false, pa, false, vb, (short)0, of[nt2], false, false);
      }
    }
  }

  // ---- normalize and store O -------------------------------------------
#pragma unroll
  for (int nt2 = 0; nt2 < 8; ++nt2)
#pragma unroll
    for (int j = 0; j < 8; ++j) {
      float v = of[nt2][j] / lstat[j];
      int r = qBase + 8 * hlf + j;
      O[((size_t)(b * TT + r)) * EMB + h * HD + nt2 * 16 + n16] = (__bf16)v;
    }
}

// ---------------------------------------------------------------------------
extern "C" void kernel_launch(void* const* d_in, const int* in_sizes, int n_in,
                              void* d_out, int out_size, void* d_ws, size_t ws_size,
                              hipStream_t stream) {
  (void)in_sizes; (void)n_in; (void)out_size; (void)ws_size;
  const float* x  = (const float*)d_in[0];
  const float* Wq = (const float*)d_in[1];
  const float* Wk = (const float*)d_in[2];
  const float* Wv = (const float*)d_in[3];
  const float* Wp = (const float*)d_in[4];
  const float* bp = (const float*)d_in[5];
  float* out = (float*)d_out;

  const int M = BB * TT;       // 4096 token rows
  const int NKV = NG * HD;     // 512 compact KV columns

  // Workspace layout (all bf16): total 38M elems = 76 MB
  __bf16* xb  = (__bf16*)d_ws;                     // (M, EMB)      16 MB
  __bf16* Wqb = xb  + (size_t)M * EMB;             // (EMB, EMB)     8 MB
  __bf16* Wpb = Wqb + (size_t)EMB * EMB;           // (EMB, EMB)     8 MB
  __bf16* Wkb = Wpb + (size_t)EMB * EMB;           // (NKV, EMB)     2 MB
  __bf16* Wvb = Wkb + (size_t)NKV * EMB;           // (NKV, EMB)     2 MB
  __bf16* Qb  = Wvb + (size_t)NKV * EMB;           // (M, EMB)      16 MB
  __bf16* Kb  = Qb  + (size_t)M * EMB;             // (M, NKV)       4 MB
  __bf16* Vb  = Kb  + (size_t)M * NKV;             // (M, NKV)       4 MB
  __bf16* Ob  = Vb  + (size_t)M * NKV;             // (M, EMB)      16 MB

  dim3 blk(256);

  // Pass 0: one-shot bf16 conversion (K/V weights compacted to head 0 per group)
  cvt_rows_f32_bf16<<<dim3(M),   blk, 0, stream>>>(x,  xb,  EMB, HD);
  cvt_rows_f32_bf16<<<dim3(EMB), blk, 0, stream>>>(Wq, Wqb, EMB, HD);
  cvt_rows_f32_bf16<<<dim3(EMB), blk, 0, stream>>>(Wp, Wpb, EMB, HD);
  cvt_rows_f32_bf16<<<dim3(NKV), blk, 0, stream>>>(Wk, Wkb, EMB, GSZ * HD);
  cvt_rows_f32_bf16<<<dim3(NKV), blk, 0, stream>>>(Wv, Wvb, EMB, GSZ * HD);

  // Pass 1: projections (pure bf16 WMMA GEMMs)
  wmma_gemm_bf16<__bf16>
      <<<dim3(EMB / 64, M / 128), blk, 0, stream>>>(xb, Wqb, Qb, nullptr, M, EMB, EMB);
  wmma_gemm_bf16<__bf16>
      <<<dim3(NKV / 64, M / 128), blk, 0, stream>>>(xb, Wkb, Kb, nullptr, M, NKV, EMB);
  wmma_gemm_bf16<__bf16>
      <<<dim3(NKV / 64, M / 128), blk, 0, stream>>>(xb, Wvb, Vb, nullptr, M, NKV, EMB);

  // Pass 2: causal GQA flash attention
  flash_attn_gqa<<<dim3(TT / 128, BB * NH), blk, 0, stream>>>(Qb, Kb, Vb, Ob);

  // Pass 3: out = Ob @ Wp^T + bp (fp32 output)
  wmma_gemm_bf16<float>
      <<<dim3(EMB / 64, M / 128), blk, 0, stream>>>(Ob, Wpb, out, bp, M, EMB, EMB);
}
